// Attention_43130061586633
// MI455X (gfx1250) — matrix operand order repaired and checked by reading, not compiled, not run on hardware
//
#include <hip/hip_runtime.h>
#include <math.h>

// ---------------------------------------------------------------- types
typedef __bf16 bf16;
typedef __attribute__((ext_vector_type(16))) __bf16 v16bf;
typedef __attribute__((ext_vector_type(8)))  __bf16 v8bf;
typedef __attribute__((ext_vector_type(8)))  float  v8f;

static constexpr int Bq = 4, Lq = 2048, Cq = 1024, Hh = 16, Dd = 64;

// W-tile LDS row stride in bf16 elements (80 bytes): keeps 16B alignment,
// 40-word stride -> at most 2-way bank reuse across the 16 column lanes.
static constexpr int WSTRIDE = 40;

__device__ inline v8f vzero8() {
  v8f z;
  #pragma unroll
  for (int i = 0; i < 8; ++i) z[i] = 0.0f;
  return z;
}

// ---- CDNA5 async global->LDS copy (ASYNCcnt), 16 bytes per lane ----------
__device__ inline void async_ld_16B(void* lds_ptr, const void* gptr) {
  const unsigned int lds_off = (unsigned int)(unsigned long long)lds_ptr;
  asm volatile("global_load_async_to_lds_b128 %0, %1, off"
               :: "v"(lds_off), "v"(gptr)
               : "memory");
}
__device__ inline void wait_async0() {
  asm volatile("s_wait_asynccnt 0x0" ::: "memory");
}

// A fragment (16x32 bf16, MxK): lane m = lane&15 holds row m.
// Lane holds K = hi*16..hi*16+15 (contiguous, matching the B fragments).
__device__ inline v16bf load_a_frag(const bf16* base, int ld, int lane) {
  const int m = lane & 15, hi = lane >> 4;
  const bf16* p = base + (size_t)m * ld;
  v8bf c0 = *(const v8bf*)(p + hi * 16);
  v8bf c1 = *(const v8bf*)(p + hi * 16 + 8);
  return __builtin_shufflevector(c0, c1, 0, 1, 2, 3, 4, 5, 6, 7,
                                          8, 9, 10, 11, 12, 13, 14, 15);
}

// B fragment (32x16 bf16, KxN) from row-major B^T rows: lane n holds column n,
// K = hi*16..hi*16+15 -> one contiguous 32B chunk.
__device__ inline v16bf load_b_frag(const bf16* base, int ld, int lane) {
  const int n = lane & 15, hi = lane >> 4;
  const bf16* p = base + (size_t)n * ld + hi * 16;
  v8bf c0 = *(const v8bf*)(p);
  v8bf c1 = *(const v8bf*)(p + 8);
  return __builtin_shufflevector(c0, c1, 0, 1, 2, 3, 4, 5, 6, 7,
                                          8, 9, 10, 11, 12, 13, 14, 15);
}

// Same B fragment out of the LDS-staged W tile (row stride WSTRIDE elems).
__device__ inline v16bf load_b_frag_lds(const bf16* base, int lane) {
  const int n = lane & 15, hi = lane >> 4;
  const bf16* p = base + n * WSTRIDE + hi * 16;
  v8bf c0 = *(const v8bf*)(p);
  v8bf c1 = *(const v8bf*)(p + 8);
  return __builtin_shufflevector(c0, c1, 0, 1, 2, 3, 4, 5, 6, 7,
                                          8, 9, 10, 11, 12, 13, 14, 15);
}

__device__ inline v8f wmma_bf16(v16bf a, v16bf b, v8f c) {
  return __builtin_amdgcn_wmma_f32_16x16x32_bf16(false, a, false, b,
                                                 (short)0, c, false, false);
}

// ---------------------------------------------------------------- fp32 -> bf16
__global__ __launch_bounds__(256) void cvt_bf16(const float* __restrict__ src,
                                                bf16* __restrict__ dst, int n) {
  for (int i = blockIdx.x * blockDim.x + threadIdx.x; i < n;
       i += gridDim.x * blockDim.x)
    dst[i] = (bf16)src[i];
}

// ---------------------------------------------------------------- GEMM
// out[m,n] = sum_k A[m,k] * W[n,k]    A:[M,K] bf16, W:[N,K] bf16 (row major)
// Block tile 128x128: 8 waves as 4 (M) x 2 (N); each wave: 32x64 = 8 WMMAs
// per 32-wide K step.  W tile (128 x 32k) staged in LDS by async DMA,
// double-buffered: next tile's copy overlaps this tile's 64 block-WMMAs.
template <bool F32OUT>
__global__ __launch_bounds__(256) void gemm_wmma(const bf16* __restrict__ A,
                                                 const bf16* __restrict__ W,
                                                 void* __restrict__ Out,
                                                 const float* __restrict__ bias,
                                                 int M, int N, int K) {
  __shared__ bf16 wtile[2][128 * WSTRIDE];   // 2 x 10 KB

  const int tid = threadIdx.x;
  const int lane = tid & 31;
  const int w = tid >> 5;
  const int mg = w & 3;            // 4 M-groups of 32 rows
  const int ng = w >> 2;           // 2 N-groups of 64 cols
  const int row0 = blockIdx.y * 128 + mg * 32;
  const int col0 = blockIdx.x * 128 + ng * 64;

  // async-copy assignment: thread -> (W row, two 16B chunks of its 64B slice)
  const int crow = tid >> 1;            // 0..127
  const int cchk = (tid & 1) * 16;      // bf16 offset of first chunk

  v8f acc[2][4];
  #pragma unroll
  for (int i = 0; i < 2; ++i)
    #pragma unroll
    for (int t = 0; t < 4; ++t) acc[i][t] = vzero8();

  const bf16* Wrow = W + (size_t)(blockIdx.x * 128 + crow) * K;
  const bf16* Arow = A + (size_t)row0 * K;

  // prologue: DMA first W tile into buffer 0
  async_ld_16B(&wtile[0][crow * WSTRIDE + cchk], Wrow + cchk);
  async_ld_16B(&wtile[0][crow * WSTRIDE + cchk + 8], Wrow + cchk + 8);

  for (int k0 = 0; k0 < K; k0 += 32) {
    const int cur = (k0 >> 5) & 1;
    wait_async0();        // this wave's slices of wtile[cur] landed
    __syncthreads();      // every wave's slices landed

    if (k0 + 32 < K) {    // DMA next tile while computing on this one
      async_ld_16B(&wtile[1 - cur][crow * WSTRIDE + cchk],
                   Wrow + k0 + 32 + cchk);
      async_ld_16B(&wtile[1 - cur][crow * WSTRIDE + cchk + 8],
                   Wrow + k0 + 32 + cchk + 8);
    }

    __builtin_prefetch(Arow + k0 + 512, 0, 1);   // global_prefetch_b8
    v16bf a0 = load_a_frag(Arow + k0, K, lane);
    v16bf a1 = load_a_frag(Arow + (size_t)16 * K + k0, K, lane);
    #pragma unroll
    for (int t = 0; t < 4; ++t) {
      v16bf b = load_b_frag_lds(&wtile[cur][(ng * 64 + 16 * t) * WSTRIDE], lane);
      acc[0][t] = wmma_bf16(a0, b, acc[0][t]);
      acc[1][t] = wmma_bf16(a1, b, acc[1][t]);
    }
    __syncthreads();      // done reading wtile[cur] before it is re-filled
  }

  const int n0 = lane & 15, hi = lane >> 4;
  #pragma unroll
  for (int i = 0; i < 2; ++i) {
    #pragma unroll
    for (int t = 0; t < 4; ++t) {
      const int col = col0 + 16 * t + n0;
      #pragma unroll
      for (int r = 0; r < 8; ++r) {
        const size_t idx = (size_t)(row0 + 16 * i + hi * 8 + r) * N + col;
        if (F32OUT)
          ((float*)Out)[idx] = acc[i][t][r] + bias[col];
        else
          ((bf16*)Out)[idx] = (bf16)acc[i][t][r];
      }
    }
  }
}

// ---------------------------------------------------------------- RoPE + split
// qkv: [B, L, 3C] bf16 -> Q,K (rotated) [B,H,L,D] bf16, V transposed [B,H,D,L]
__global__ __launch_bounds__(256) void rope_split(const bf16* __restrict__ qkv,
                                                  const float* __restrict__ cosb,
                                                  const float* __restrict__ sinb,
                                                  bf16* __restrict__ Qm,
                                                  bf16* __restrict__ Km,
                                                  bf16* __restrict__ Vt) {
  const int idx = blockIdx.x * blockDim.x + threadIdx.x;
  if (idx >= Bq * Lq * Hh) return;
  const int h = idx % Hh;
  const int t = idx / Hh;
  const int l = t % Lq;
  const int b = t / Lq;

  const bf16* base = qkv + (size_t)(b * Lq + l) * (3 * Cq);
  const bf16* qp = base + h * Dd;
  const bf16* kp = base + Cq + h * Dd;
  const bf16* vp = base + 2 * Cq + h * Dd;
  const float* cr = cosb + l * (Dd / 2);
  const float* sr = sinb + l * (Dd / 2);

  const size_t qkbase = ((size_t)(b * Hh + h) * Lq + l) * Dd;
  #pragma unroll 4
  for (int i = 0; i < Dd / 2; ++i) {
    const float c = cr[i], s = sr[i];
    const float qe = (float)qp[2 * i], qo = (float)qp[2 * i + 1];
    Qm[qkbase + 2 * i]     = (bf16)(qe * c - qo * s);
    Qm[qkbase + 2 * i + 1] = (bf16)(qe * s + qo * c);
    const float ke = (float)kp[2 * i], ko = (float)kp[2 * i + 1];
    Km[qkbase + 2 * i]     = (bf16)(ke * c - ko * s);
    Km[qkbase + 2 * i + 1] = (bf16)(ke * s + ko * c);
  }
  const size_t vtb = ((size_t)(b * Hh + h) * Dd) * Lq + l;
  #pragma unroll 8
  for (int d = 0; d < Dd; ++d) Vt[vtb + (size_t)d * Lq] = vp[d];
}

// ---------------------------------------------------------------- flash attn
// Transposed formulation: S^T = K Q^T and O^T = V^T P^T, so each LANE owns one
// query column -> softmax stats are per-lane VALU + two xor-16 shuffles per
// 32-key tile.  One wave per 16-query tile, 4 waves/block.  O: [B, L, C] bf16.
__global__ __launch_bounds__(128) void flash_attn(const bf16* __restrict__ Qm,
                                                  const bf16* __restrict__ Km,
                                                  const bf16* __restrict__ Vt,
                                                  bf16* __restrict__ O) {
  __shared__ bf16 pbuf[4][16 * 32];   // per-wave P^T staging: [query][key]

  const int lane = threadIdx.x & 31;
  const int w = threadIdx.x >> 5;
  const int bh = blockIdx.y;
  const int b = bh / Hh, h = bh % Hh;
  const int q0 = blockIdx.x * 64 + w * 16;
  const int n0 = lane & 15, hi = lane >> 4;

  const bf16* Qb = Qm + ((size_t)bh * Lq + q0) * Dd;
  const bf16* Kb = Km + (size_t)bh * Lq * Dd;
  const bf16* Vb = Vt + (size_t)bh * Dd * Lq;

  // Q as loop-invariant B operand of S^T = K * Q^T (lane n = query q0+n)
  const v16bf bq0 = load_b_frag(Qb, Dd, lane);        // d = 0..31
  const v16bf bq1 = load_b_frag(Qb + 32, Dd, lane);   // d = 32..63

  float mq = -1e30f, lq = 0.0f;   // per-lane running max / sum for query n0
  v8f acc[4];                     // O^T: d-tile t, rows d=16t+hi*8+r, col n0
  #pragma unroll
  for (int t = 0; t < 4; ++t) acc[t] = vzero8();

  const float scale = 0.125f;  // D^-0.5, D = 64

  for (int kb = 0; kb < Lq; kb += 32) {
    // ---- S^T[key, query]: keys kb..+15 in st0, kb+16..+31 in st1
    const bf16* krow = Kb + (size_t)kb * Dd;
    v8f st0 = vzero8(), st1 = vzero8();
    st0 = wmma_bf16(load_a_frag(krow, Dd, lane), bq0, st0);
    st0 = wmma_bf16(load_a_frag(krow + 32, Dd, lane), bq1, st0);
    const bf16* krow2 = krow + 16 * Dd;
    st1 = wmma_bf16(load_a_frag(krow2, Dd, lane), bq0, st1);
    st1 = wmma_bf16(load_a_frag(krow2 + 32, Dd, lane), bq1, st1);

    // ---- per-lane online softmax over this tile's 32 keys
    float pm = -1e30f;
    #pragma unroll
    for (int r = 0; r < 8; ++r)
      pm = fmaxf(pm, fmaxf(st0[r] * scale, st1[r] * scale));
    pm = fmaxf(pm, __shfl_xor(pm, 16, 32));          // join key halves
    const float mnew = fmaxf(mq, pm);
    const float alpha = __expf(mq - mnew);
    mq = mnew;

    float psum = 0.0f;
    v8bf p0v, p1v;
    #pragma unroll
    for (int r = 0; r < 8; ++r) {
      const float p0 = __expf(st0[r] * scale - mnew);
      const float p1 = __expf(st1[r] * scale - mnew);
      psum += p0 + p1;
      p0v[r] = (bf16)p0;
      p1v[r] = (bf16)p1;
    }
    psum += __shfl_xor(psum, 16, 32);
    lq = lq * alpha + psum;
    #pragma unroll
    for (int t = 0; t < 4; ++t)
      #pragma unroll
      for (int r = 0; r < 8; ++r) acc[t][r] *= alpha;

    // ---- stage P^T as [query][key] rows (two 16B LDS stores per lane)
    *(v8bf*)&pbuf[w][n0 * 32 + hi * 8]      = p0v;   // keys kb+hi*8+r
    *(v8bf*)&pbuf[w][n0 * 32 + 16 + hi * 8] = p1v;   // keys kb+16+hi*8+r
    __syncthreads();

    // ---- O^T += V^T_tile * P^T  (B operand: lane n = query, K = 32 keys)
    v16bf bp;
    {
      const bf16* p = &pbuf[w][n0 * 32 + hi * 16];
      v8bf c0 = *(const v8bf*)(p);
      v8bf c1 = *(const v8bf*)(p + 8);
      bp = __builtin_shufflevector(c0, c1, 0, 1, 2, 3, 4, 5, 6, 7,
                                           8, 9, 10, 11, 12, 13, 14, 15);
    }
    #pragma unroll
    for (int t = 0; t < 4; ++t) {
      v16bf av = load_a_frag(Vb + (size_t)(16 * t) * Lq + kb, Lq, lane);
      acc[t] = wmma_bf16(av, bp, acc[t]);
    }
    __syncthreads();   // protect pbuf before next iteration overwrites
  }

  // ---- epilogue: O[b, q0+n0, h*D + 16t + hi*8 + r] -- contiguous in r,
  // so one 16B store per d-tile per lane.
  const float inv_l = 1.0f / lq;
  bf16* orow = O + ((size_t)b * Lq + q0 + n0) * Cq + h * Dd;
  #pragma unroll
  for (int t = 0; t < 4; ++t) {
    v8bf ov;
    #pragma unroll
    for (int r = 0; r < 8; ++r) ov[r] = (bf16)(acc[t][r] * inv_l);
    *(v8bf*)(orow + 16 * t + hi * 8) = ov;
  }
}

// ---------------------------------------------------------------- launch
extern "C" void kernel_launch(void* const* d_in, const int* in_sizes, int n_in,
                              void* d_out, int out_size, void* d_ws, size_t ws_size,
                              hipStream_t stream) {
  const float* x      = (const float*)d_in[0];
  const float* cosb   = (const float*)d_in[1];
  const float* sinb   = (const float*)d_in[2];
  const float* w_qkv  = (const float*)d_in[3];
  const float* w_proj = (const float*)d_in[4];
  const float* b_proj = (const float*)d_in[5];

  const int M = Bq * Lq;          // 8192
  char* ws = (char*)d_ws;
  bf16* xb   = (bf16*)ws;                      ws += (size_t)M * Cq * 2;        // 16 MB
  bf16* wq   = (bf16*)ws;                      ws += (size_t)3 * Cq * Cq * 2;   //  6 MB
  bf16* wp   = (bf16*)ws;                      ws += (size_t)Cq * Cq * 2;       //  2 MB
  bf16* qkv  = (bf16*)ws;                      ws += (size_t)M * 3 * Cq * 2;    // 48 MB
  bf16* Qb   = (bf16*)ws;                      ws += (size_t)M * Cq * 2;        // 16 MB
  bf16* Kb   = (bf16*)ws;                      ws += (size_t)M * Cq * 2;        // 16 MB
  bf16* Vt   = (bf16*)ws;                      ws += (size_t)M * Cq * 2;        // 16 MB
  bf16* Ob   = (bf16*)ws;                                                      // 16 MB

  cvt_bf16<<<512, 256, 0, stream>>>(x, xb, M * Cq);
  cvt_bf16<<<256, 256, 0, stream>>>(w_qkv, wq, 3 * Cq * Cq);
  cvt_bf16<<<128, 256, 0, stream>>>(w_proj, wp, Cq * Cq);

  // QKV projection: [8192,1024] x [3072,1024]^T -> [8192,3072]
  gemm_wmma<false><<<dim3(3 * Cq / 128, M / 128), 256, 0, stream>>>(
      xb, wq, qkv, nullptr, M, 3 * Cq, Cq);

  rope_split<<<(Bq * Lq * Hh + 255) / 256, 256, 0, stream>>>(qkv, cosb, sinb,
                                                             Qb, Kb, Vt);

  flash_attn<<<dim3(Lq / 64, Bq * Hh), 128, 0, stream>>>(Qb, Kb, Vt, Ob);

  // Output projection: [8192,1024] x [1024,1024]^T + bias -> f32 d_out
  gemm_wmma<true><<<dim3(Cq / 128, M / 128), 256, 0, stream>>>(
      Ob, wp, d_out, b_proj, M, Cq, Cq);
}
